// FieldLoss_43319040147845
// MI455X (gfx1250) — compile-verified, matrix operand order checked by reading, not run on hardware
//
#include <hip/hip_runtime.h>
#include <hip/hip_bf16.h>
#include <cstdint>

typedef _Float16 v16h __attribute__((ext_vector_type(16)));
typedef _Float16 h8   __attribute__((ext_vector_type(8)));
typedef float    v8f  __attribute__((ext_vector_type(8)));

#define HH   256
#define WW   256
#define HW   65536
#define CC   21            // seg/mask channels
#define NFG  20
#define KTOP 100
#define KP   112           // KTOP padded to 7*16
#define EM   32            // 21 padded to 32 for WMMA contraction
#define TS   16

// ---------------------------------------------------------------- utilities
__global__ void k_zero32(uint32_t* p, size_t n) {
    size_t i = (size_t)blockIdx.x * blockDim.x + threadIdx.x;
    size_t stride = (size_t)gridDim.x * blockDim.x;
    for (; i < n; i += stride) p[i] = 0u;
}
__global__ void k_zero_out(float* p, int n) {
    int i = blockIdx.x * blockDim.x + threadIdx.x;
    if (i < n) p[i] = 0.f;
}

// --------------------------------------------- 1) softmax(seg_map*1000)[1:]
__global__ void k_seg_softmax(const float* __restrict__ segmap,
                              float* __restrict__ seg, int n) {
    int i = blockIdx.x * blockDim.x + threadIdx.x;
    if (i >= n * HW) return;
    int b = i / HW, p = i % HW;
    const float* src = segmap + (size_t)b * CC * HW + p;
    float z[CC]; float mx = -3.0e38f;
    #pragma unroll
    for (int c = 0; c < CC; ++c) { float v = src[(size_t)c * HW] * 1000.f; z[c] = v; mx = fmaxf(mx, v); }
    float s = 0.f;
    #pragma unroll
    for (int c = 0; c < CC; ++c) { z[c] = __expf(z[c] - mx); s += z[c]; }
    float inv = 1.f / s;
    float* dst = seg + (size_t)b * NFG * HW + p;
    #pragma unroll
    for (int c = 1; c < CC; ++c) dst[(size_t)(c - 1) * HW] = z[c] * inv;
}

// --------------------------- 2) 5x5 Sobel (cross-correlation) -> mag key, q
__device__ __forceinline__ int quant_oct(float o) {
    const float d = 3.1416f / 8.0f;
    if (o >=  d      && o <  3.f*d) return 0;
    if (o >=  3.f*d  && o <  5.f*d) return 1;
    if (o >=  5.f*d  && o <  7.f*d) return 2;
    if ((o >= 7.f*d  && o <  8.f*d) || (o >= -8.f*d && o < -7.f*d)) return 3;
    if (o >= -7.f*d  && o < -5.f*d) return 4;
    if (o >= -5.f*d  && o < -3.f*d) return 5;
    if (o >= -3.f*d  && o < -d)     return 6;
    if (o >= -d      && o <  d)     return 7;
    return 3;
}

__global__ void k_sobel(const float* __restrict__ seg, const float* __restrict__ label,
                        uint32_t* __restrict__ key, int8_t* __restrict__ qv) {
    const float GX[5][5] = {{2,1,1e-6f,-1,-2},{3,2,1e-6f,-2,-3},{4,3,0,-3,-4},
                            {3,2,1e-6f,-2,-3},{2,1,1e-6f,-1,-2}};
    const float GY[5][5] = {{2,3,4,3,2},{1,2,3,2,1},{1e-6f,1e-6f,1e-6f,1e-6f,1e-6f},
                            {-1,-2,-3,-2,-1},{-2,-3,-4,-3,-2}};
    int img = blockIdx.z;               // b*NFG+cf
    int b = img / NFG, cf = img % NFG;
    float lab = label[b * CC + 1 + cf];
    const float* S = seg + (size_t)img * HW;
    __shared__ float tile[TS + 4][TS + 4];
    int tx = threadIdx.x, ty = threadIdx.y;
    int x0 = blockIdx.x * TS, y0 = blockIdx.y * TS;
    for (int i = ty * TS + tx; i < (TS + 4) * (TS + 4); i += TS * TS) {
        int ly = i / (TS + 4), lx = i % (TS + 4);
        int gx = x0 + lx - 2, gy = y0 + ly - 2;
        tile[ly][lx] = (gx >= 0 && gx < WW && gy >= 0 && gy < HH) ? S[(size_t)gy * WW + gx] : 0.f;
    }
    __syncthreads();
    float sx = 0.f, sy = 0.f;
    #pragma unroll
    for (int dy = 0; dy < 5; ++dy)
        #pragma unroll
        for (int dx = 0; dx < 5; ++dx) {
            float v = tile[ty + dy][tx + dx];
            sx += v * GX[dy][dx]; sy += v * GY[dy][dx];
        }
    sx *= lab; sy *= lab;
    float mag = sqrtf(sx * sx + sy * sy + 1e-8f);
    int q = quant_oct(atan2f(sy, sx));
    size_t o = (size_t)img * HW + (size_t)(y0 + ty) * WW + (x0 + tx);
    key[o] = __float_as_uint(mag);      // mag > 0 -> order-preserving
    qv[o]  = (int8_t)q;
}

// ----------------------------- 3) per-row top-K via radix select (ties: low idx)
__global__ void k_topk(const uint32_t* __restrict__ keys, int* __restrict__ pos) {
    int row = blockIdx.x;
    const uint32_t* kr = keys + (size_t)row * HW;
    __shared__ uint32_t hist[256];
    __shared__ uint32_t s_prefix;
    __shared__ int s_want, s_cnt;
    __shared__ uint8_t flag[256];
    int t = threadIdx.x;
    if (t == 0) { s_prefix = 0u; s_want = KTOP; }
    __syncthreads();
    for (int pass = 0; pass < 4; ++pass) {
        int shift = 24 - 8 * pass;
        hist[t] = 0u;
        __syncthreads();
        uint32_t pref = s_prefix;
        uint32_t himask = (pass == 0) ? 0u : (0xFFFFFFFFu << (shift + 8));
        for (int i = t; i < HW; i += 256) {
            if (i + 2048 < HW) __builtin_prefetch(&kr[i + 2048], 0, 1);
            uint32_t kv = kr[i];
            if ((kv & himask) == (pref & himask))
                atomicAdd(&hist[(kv >> shift) & 255u], 1u);
        }
        __syncthreads();
        if (t == 0) {
            int cum = 0, want = s_want, digit = 0;
            for (int bn = 255; bn >= 0; --bn) {
                int c = (int)hist[bn];
                if (cum + c >= want) { digit = bn; s_want = want - cum; break; }
                cum += c;
            }
            s_prefix = pref | ((uint32_t)digit << shift);
        }
        __syncthreads();
    }
    uint32_t T = s_prefix;              // K-th largest key
    if (t == 0) s_cnt = 0;
    __syncthreads();
    int* prow = pos + row * KTOP;
    for (int i = t; i < HW; i += 256) {
        if (kr[i] > T) { int a = atomicAdd(&s_cnt, 1); if (a < KTOP) prow[a] = i; }
    }
    __syncthreads();
    for (int base = 0; base < HW && s_cnt < KTOP; base += 256) {
        flag[t] = (kr[base + t] == T) ? 1 : 0;
        __syncthreads();
        if (t == 0) {
            for (int j = 0; j < 256 && s_cnt < KTOP; ++j)
                if (flag[j]) prow[s_cnt++] = base + j;
        }
        __syncthreads();
    }
}

// ------------------------------------- 4) per-selection index/valid compute
__global__ void k_indices(const int* __restrict__ pos, const int8_t* __restrict__ qv,
                          int* __restrict__ oi, int* __restrict__ ii,
                          float* __restrict__ vout, float* __restrict__ vin, int total) {
    int idx = blockIdx.x * blockDim.x + threadIdx.x;
    if (idx >= total) return;
    int row = idx / KTOP;
    int p = pos[idx];
    int o = (int)qv[(size_t)row * HW + p] + 1;        // 1..8
    int mod4  = (o % 4 == 0) ? 1 : 0;
    int horiz = (o == 2 || o == 6) ? 1 : 0;
    int p_out = (o < 4) ? 25 : -5;
    int p_in  = (o < 4) ? -5 : 1;
    int s_out = (o % 2 == 0) ? -1 : 1;
    int s_in  = -s_out;
    int out_idx = p + p_out * mod4 * WW + s_out * horiz;
    int in_idx  = p + p_in  * mod4 * WW + s_in  * horiz;
    auto validf = [](int x) -> float {
        int r = x % (WW - 1); if (r < 0) r += (WW - 1);     // jnp.remainder semantics
        return (r != 0 && r != 1 && x > 0 && x < HW - 1) ? 1.f : 0.f;
    };
    vout[idx] = validf(out_idx);
    vin[idx]  = validf(in_idx);
    oi[idx] = min(max(out_idx, 0), HW - 1);
    ii[idx] = min(max(in_idx,  0), HW - 1);
}

// ---------------- 5) gather columns + channel softmax -> f16 WMMA operands
__global__ void k_gather_dense(const float* __restrict__ dense,
                               const int* __restrict__ oi, const int* __restrict__ ii,
                               _Float16* __restrict__ outs, _Float16* __restrict__ ins,
                               int total, int CH) {
    int g = blockIdx.x;
    int e = (g >= total) ? 1 : 0;
    int idx = e ? (g - total) : g;
    int row = idx / KTOP, k = idx % KTOP;
    int b = row / NFG;
    int col = (e ? ii : oi)[idx];
    const float* src = dense + (size_t)b * CH * HW + col;
    int t = threadIdx.x;                 // == channel, blockDim == CH == 256
    float x = src[(size_t)t * HW];
    __shared__ float red[256];
    red[t] = x; __syncthreads();
    for (int s = 128; s > 0; s >>= 1) { if (t < s) red[t] = fmaxf(red[t], red[t + s]); __syncthreads(); }
    float mx = red[0]; __syncthreads();
    float ex = __expf(x - mx);
    red[t] = ex; __syncthreads();
    for (int s = 128; s > 0; s >>= 1) { if (t < s) red[t] += red[t + s]; __syncthreads(); }
    float sm = ex / red[0];
    _Float16* dst = (e ? ins : outs) + ((size_t)row * KP + k) * CH + t;
    *dst = (_Float16)sm;
}

__global__ void k_gather_mask(const float* __restrict__ msk,
                              const int* __restrict__ oi, const int* __restrict__ ii,
                              _Float16* __restrict__ outsm, _Float16* __restrict__ insm,
                              int total) {
    int g = blockIdx.x;
    int e = (g >= total) ? 1 : 0;
    int idx = e ? (g - total) : g;
    int row = idx / KTOP, k = idx % KTOP;
    int b = row / NFG;
    int col = (e ? ii : oi)[idx];
    const float* src = msk + (size_t)b * CC * HW + col;
    int t = threadIdx.x;                 // blockDim == 32
    float x = (t < CC) ? src[(size_t)t * HW] : -3.0e38f;
    __shared__ float red[32];
    red[t] = x; __syncthreads();
    for (int s = 16; s > 0; s >>= 1) { if (t < s) red[t] = fmaxf(red[t], red[t + s]); __syncthreads(); }
    float mx = red[0]; __syncthreads();
    float ex = (t < CC) ? __expf(x - mx) : 0.f;
    red[t] = ex; __syncthreads();
    for (int s = 16; s > 0; s >>= 1) { if (t < s) red[t] += red[t + s]; __syncthreads(); }
    if (t < CC) {
        _Float16* dst = (e ? insm : outsm) + ((size_t)row * KP + k) * EM + t;
        *dst = (_Float16)(ex / red[0]);  // lanes 21..31 stay zero (pre-zeroed)
    }
}

// ------------------------- 6) batched sim = A(KPxKD) * B(KPxKD)^T via WMMA
__global__ void k_wmma_gemm(const _Float16* __restrict__ A, const _Float16* __restrict__ B,
                            float* __restrict__ D, int KD) {
    int mat = blockIdx.x;
    int wave = (threadIdx.x >> 5) + (blockIdx.y << 3);
    const int NTM = KP / 16;                 // 7
    if (wave >= NTM * NTM) return;
    int tm = wave / NTM, tn = wave % NTM;
    int lane = threadIdx.x & 31;
    int half = lane >> 4, r16 = lane & 15;
    // A: lane(0-15)->M=r16 K 0-7/16-23 ; lane(16-31)->M=r16 K 8-15/24-31 (ISA 16-bit A layout)
    const _Float16* Ab = A + (size_t)mat * KP * KD + (size_t)(tm * 16 + r16) * KD + half * 8;
    const _Float16* Bb = B + (size_t)mat * KP * KD + (size_t)(tn * 16 + r16) * KD + half * 8;
    v8f acc = {};
    for (int kk = 0; kk < KD; kk += 32) {
        h8 alo = *(const h8*)(Ab + kk);
        h8 ahi = *(const h8*)(Ab + kk + 16);
        h8 blo = *(const h8*)(Bb + kk);
        h8 bhi = *(const h8*)(Bb + kk + 16);
        v16h a, bv;
        #pragma unroll
        for (int i = 0; i < 8; ++i) { a[i] = alo[i]; a[8 + i] = ahi[i]; bv[i] = blo[i]; bv[8 + i] = bhi[i]; }
        acc = __builtin_amdgcn_wmma_f32_16x16x32_f16(false, a, false, bv, (short)0, acc, false, false);
    }
    // D layout: VGPR r -> M = r + 8*half, N = r16
    float* Drow = D + (size_t)mat * KP * KP;
    #pragma unroll
    for (int r = 0; r < 8; ++r)
        Drow[(size_t)(tm * 16 + r + half * 8) * KP + (tn * 16 + r16)] = acc[r];
}

// ------------------------------------ 7) means, signs, beacon, gate, reduce
__global__ void k_finalize(const float* __restrict__ sim, const float* __restrict__ simm,
                           const float* __restrict__ vout, const float* __restrict__ vin,
                           const float* __restrict__ label, float* __restrict__ out) {
    int row = blockIdx.x;                    // b*NFG+cf
    int b = row / NFG, cf = row % NFG;
    const float* S  = sim  + (size_t)row * KP * KP;
    const float* Sm = simm + (size_t)row * KP * KP;
    __shared__ float fvo[KTOP], fvi[KTOP];
    __shared__ float rm[KTOP], cm[KTOP], rmm[KTOP], cmm[KTOP];
    __shared__ float s_svo, s_svi, s_g, s_gm;
    int t = threadIdx.x;
    if (t < KTOP) { fvo[t] = vout[row * KTOP + t]; fvi[t] = vin[row * KTOP + t]; }
    __syncthreads();
    if (t == 0) {
        float a = 0.f, c = 0.f;
        for (int i = 0; i < KTOP; ++i) { a += fvo[i]; c += fvi[i]; }
        s_svo = a; s_svi = c;
    }
    __syncthreads();
    float svo = s_svo, svi = s_svi;
    float ci = fmaxf(svi, 1.f), co = fmaxf(svo, 1.f);
    if (t < KTOP) {
        float a = 0.f, bm = 0.f, c = 0.f, dm = 0.f;
        for (int j = 0; j < KTOP; ++j) { a += S[t * KP + j] * fvi[j]; bm += Sm[t * KP + j] * fvi[j]; }
        rm[t] = a / ci; rmm[t] = bm / ci;
        for (int i = 0; i < KTOP; ++i) { c += S[i * KP + t] * fvo[i]; dm += Sm[i * KP + t] * fvo[i]; }
        cm[t] = c / co; cmm[t] = dm / co;
    }
    __syncthreads();
    if (t == 0) {
        float gn = 0.f, gmn = 0.f;
        for (int i = 0; i < KTOP; ++i) { gn += fvo[i] * rm[i] * ci; gmn += fvo[i] * rmm[i] * ci; }
        float den = fmaxf(svo * svi, 1.f);
        s_g = gn / den; s_gm = gmn / den;
    }
    __syncthreads();
    if (t == 0) {
        float g = s_g, gm = s_gm, loss = 0.f;
        for (int pass = 0; pass < 2; ++pass) {
            const float* vals  = pass ? cm  : rm;
            const float* valsm = pass ? cmm : rmm;
            const float* vv    = pass ? fvi : fvo;
            float sum[4] = {0, 0, 0, 0}, cnt[4] = {0, 0, 0, 0};
            for (int i = 0; i < KTOP; ++i) {
                bool smk = valsm[i] > gm;      // sgn_mask
                bool sdn = vals[i]  > g;       // sgn_dense
                int cat = smk ? (sdn ? 3 : 0) : (sdn ? 1 : 2); // FP,FN,TN,TP = 0,1,2,3
                if (vv[i] > 0.5f) { sum[cat] += vals[i]; cnt[cat] += 1.f; }
            }
            const float sgn[4] = {-1.f, 1.f, 1.f, -1.f};
            for (int c4 = 0; c4 < 4; ++c4)
                loss += sgn[c4] * (cnt[c4] > 0.f ? sum[c4] / fmaxf(cnt[c4], 1.f) : 0.f);
        }
        float lab = label[b * CC + 1 + cf];
        float gate = ((svi >= 10.f) && (svo >= 10.f) && (lab > 0.f)) ? 1.f : 0.f;
        atomicAdd(out, loss * gate);
    }
}

// --------------------------------------------------------------- launcher
extern "C" void kernel_launch(void* const* d_in, const int* in_sizes, int n_in,
                              void* d_out, int out_size, void* d_ws, size_t ws_size,
                              hipStream_t stream) {
    const float* seg_map  = (const float*)d_in[0];
    const float* dense_ft = (const float*)d_in[1];
    const float* mask     = (const float*)d_in[2];
    const float* label    = (const float*)d_in[3];
    float* out = (float*)d_out;

    const int n  = in_sizes[0] / (CC * HW);          // 2
    const int CH = in_sizes[1] / (n * HW);           // 256
    const int ROWS = n * NFG;                        // 40
    const int TOTAL = ROWS * KTOP;                   // 4000

    char* ws = (char*)d_ws;
    size_t off = 0;
    auto alloc = [&](size_t bytes) { size_t o = off; off = (off + bytes + 255) & ~(size_t)255; return o; };
    size_t o_seg  = alloc((size_t)ROWS * HW * 4);
    size_t o_key  = alloc((size_t)ROWS * HW * 4);
    size_t o_qv   = alloc((size_t)ROWS * HW);
    size_t o_pos  = alloc((size_t)TOTAL * 4);
    size_t o_oi   = alloc((size_t)TOTAL * 4);
    size_t o_ii   = alloc((size_t)TOTAL * 4);
    size_t o_vout = alloc((size_t)TOTAL * 4);
    size_t o_vin  = alloc((size_t)TOTAL * 4);
    size_t h16    = alloc(0);                        // start of f16 region (contiguous)
    size_t o_outs  = alloc((size_t)ROWS * KP * CH * 2);
    size_t o_ins   = alloc((size_t)ROWS * KP * CH * 2);
    size_t o_outsm = alloc((size_t)ROWS * KP * EM * 2);
    size_t o_insm  = alloc((size_t)ROWS * KP * EM * 2);
    size_t h16_end = alloc(0);
    size_t o_sim  = alloc((size_t)ROWS * KP * KP * 4);
    size_t o_simm = alloc((size_t)ROWS * KP * KP * 4);
    (void)ws_size;

    float*    seg  = (float*)(ws + o_seg);
    uint32_t* key  = (uint32_t*)(ws + o_key);
    int8_t*   qv   = (int8_t*)(ws + o_qv);
    int*      pos  = (int*)(ws + o_pos);
    int*      oi   = (int*)(ws + o_oi);
    int*      ii   = (int*)(ws + o_ii);
    float*    vout = (float*)(ws + o_vout);
    float*    vin  = (float*)(ws + o_vin);
    _Float16* outs  = (_Float16*)(ws + o_outs);
    _Float16* ins   = (_Float16*)(ws + o_ins);
    _Float16* outsm = (_Float16*)(ws + o_outsm);
    _Float16* insm  = (_Float16*)(ws + o_insm);
    float* sim  = (float*)(ws + o_sim);
    float* simm = (float*)(ws + o_simm);

    // 0) zero output + f16 operand region (rows >= K and e >= 21 must be 0)
    size_t nz = (h16_end - h16) / 4;
    k_zero32<<<1024, 256, 0, stream>>>((uint32_t*)(ws + h16), nz);
    k_zero_out<<<1, 32, 0, stream>>>(out, out_size);

    // 1) seg softmax
    k_seg_softmax<<<(n * HW + 255) / 256, 256, 0, stream>>>(seg_map, seg, n);

    // 2) sobel -> mag keys + quantized orientation
    k_sobel<<<dim3(WW / TS, HH / TS, ROWS), dim3(TS, TS), 0, stream>>>(seg, label, key, qv);

    // 3) top-K per row
    k_topk<<<ROWS, 256, 0, stream>>>(key, pos);

    // 4) indices + validity
    k_indices<<<(TOTAL + 255) / 256, 256, 0, stream>>>(pos, qv, oi, ii, vout, vin, TOTAL);

    // 5) gather + per-column softmax into f16 WMMA operands
    k_gather_dense<<<2 * TOTAL, CH, 0, stream>>>(dense_ft, oi, ii, outs, ins, TOTAL, CH);
    k_gather_mask<<<2 * TOTAL, 32, 0, stream>>>(mask, oi, ii, outsm, insm, TOTAL);

    // 6) batched WMMA GEMMs: sim (K=256) and sim_m (K=32, zero padded 21->32)
    k_wmma_gemm<<<dim3(ROWS, 7), 256, 0, stream>>>(outs, ins, sim, CH);
    k_wmma_gemm<<<dim3(ROWS, 7), 256, 0, stream>>>(outsm, insm, simm, EM);

    // 7) reductions + beacon + gate -> scalar
    k_finalize<<<ROWS, 128, 0, stream>>>(sim, simm, vout, vin, label, out);
}
